// MultiHeadAttention_44710609551804
// MI455X (gfx1250) — matrix operand order repaired and checked by reading, not compiled, not run on hardware
//
#include <hip/hip_runtime.h>
#include <hip/hip_bf16.h>
#include <math.h>

typedef __attribute__((ext_vector_type(16))) _Float16 v16h;
typedef __attribute__((ext_vector_type(8)))  _Float16 v8h;
typedef __attribute__((ext_vector_type(8)))  float    v8f;
typedef __attribute__((ext_vector_type(4)))  float    v4f;

#define S_LEN 2048
#define D_DIM 512
#define H_NUM 8
#define DK 64
#define BATCH 4
#define ROWS (BATCH * S_LEN)          // 8192
#define WELEM (D_DIM * D_DIM)         // 262144

// ---------------- WMMA helpers (gfx1250 layouts, wave32) ----------------

__device__ __forceinline__ v8f wmma_f16(v16h a, v16h b, v8f c) {
    return __builtin_amdgcn_wmma_f32_16x16x32_f16(
        /*neg_a=*/false, a, /*neg_b=*/false, b,
        /*c_mod=*/(short)0, c, /*reuse_a=*/false, /*reuse_b=*/false);
}

// A fragment: 16x32 f16 row-major, ld in elements. Per lane this is two
// contiguous 16B chunks at +8*kh and +16+8*kh -> vector loads.
__device__ __forceinline__ v16h load_a_f16(const _Float16* src, int ld) {
    int lane = threadIdx.x & 31;
    int m = lane & 15;
    int kh = lane >> 4;
    const _Float16* p = src + (size_t)m * ld + 8 * kh;
    v8h lo = *(const v8h*)(p);
    v8h hi = *(const v8h*)(p + 16);
    return __builtin_shufflevector(lo, hi, 0, 1, 2, 3, 4, 5, 6, 7,
                                   8, 9, 10, 11, 12, 13, 14, 15);
}

// A fragment from f32 LDS source (probabilities), converted to f16.
__device__ __forceinline__ v16h load_a_f32(const float* src, int ld) {
    int lane = threadIdx.x & 31;
    int m = lane & 15;
    int kh = lane >> 4;
    const float* p = src + (size_t)m * ld + 8 * kh;
    v4f f0 = *(const v4f*)(p);
    v4f f1 = *(const v4f*)(p + 4);
    v4f f2 = *(const v4f*)(p + 16);
    v4f f3 = *(const v4f*)(p + 20);
    v16h a;
#pragma unroll
    for (int i = 0; i < 4; ++i) {
        a[i]      = (_Float16)f0[i];
        a[4 + i]  = (_Float16)f1[i];
        a[8 + i]  = (_Float16)f2[i];
        a[12 + i] = (_Float16)f3[i];
    }
    return a;
}

// B fragment 32x16 from a TRANSPOSED source laid out [n][k] (row-major, ld =
// stride between n-rows). Each lane reads two 16B chunks at +8*kg and +16+8*kg.
__device__ __forceinline__ v16h load_bt_f16(const _Float16* srcT, int ld) {
    int lane = threadIdx.x & 31;
    int n = lane & 15;
    int kg = lane >> 4;
    const _Float16* p = srcT + (size_t)n * ld + kg * 8;
    v8h lo = *(const v8h*)(p);
    v8h hi = *(const v8h*)(p + 16);
    return __builtin_shufflevector(lo, hi, 0, 1, 2, 3, 4, 5, 6, 7,
                                   8, 9, 10, 11, 12, 13, 14, 15);
}

// ---------------- Stage 1: LayerNorm -> f16 ----------------
__global__ void ln_f16_kernel(const float* __restrict__ q,
                              const float* __restrict__ k,
                              const float* __restrict__ v,
                              const float* __restrict__ g,
                              const float* __restrict__ lnb,
                              _Float16* __restrict__ xn) {
    int t = blockIdx.y;
    const float* x = (t == 0) ? q : ((t == 1) ? k : v);
    int row = blockIdx.x * 4 + (threadIdx.x >> 5);
    int lane = threadIdx.x & 31;
    const float* xr = x + (size_t)row * D_DIM;

    float vals[16];
    float s = 0.f;
#pragma unroll
    for (int i = 0; i < 16; ++i) { vals[i] = xr[lane + 32 * i]; s += vals[i]; }
#pragma unroll
    for (int off = 16; off > 0; off >>= 1) s += __shfl_xor(s, off, 32);
    float mu = s * (1.0f / D_DIM);
    float vs = 0.f;
#pragma unroll
    for (int i = 0; i < 16; ++i) { float d = vals[i] - mu; vs += d * d; }
#pragma unroll
    for (int off = 16; off > 0; off >>= 1) vs += __shfl_xor(vs, off, 32);
    float rs = rsqrtf(vs * (1.0f / D_DIM) + 1e-5f);

    _Float16* o = xn + ((size_t)t * ROWS + row) * D_DIM;
#pragma unroll
    for (int i = 0; i < 16; ++i) {
        int d = lane + 32 * i;
        o[d] = (_Float16)((vals[i] - mu) * rs * g[d] + lnb[d]);
    }
}

// ---------------- Stage 2: weights f32 -> f16 TRANSPOSED ----------------
// w16t[mat][n][k] = w[mat][k][n]  (so B-fragments are contiguous per lane)
__global__ void wcvt_kernel(const float* __restrict__ wq,
                            const float* __restrict__ wk,
                            const float* __restrict__ wv,
                            const float* __restrict__ wo,
                            _Float16* __restrict__ w16t) {
    int i = blockIdx.x * blockDim.x + threadIdx.x;   // 4*262144 threads
    int mat = i >> 18;
    int j = i & (WELEM - 1);
    int n = j >> 9;
    int k = j & (D_DIM - 1);
    const float* src = (mat == 0) ? wq : (mat == 1) ? wk : (mat == 2) ? wv : wo;
    w16t[i] = (_Float16)src[(size_t)k * D_DIM + n];
}

// ---------------- Stage 3: QKV projections (WMMA) ----------------
// Q,K -> head layout [bh][s][dk] ; V -> transposed [bh][dk][s]
__global__ __launch_bounds__(256) void gemm_qkv_kernel(
        const _Float16* __restrict__ xn, const _Float16* __restrict__ w16t,
        const float* __restrict__ bq, const float* __restrict__ bk,
        const float* __restrict__ bv,
        _Float16* __restrict__ hq, _Float16* __restrict__ hk,
        _Float16* __restrict__ vT) {
    int t = blockIdx.z;
    const _Float16* A  = xn   + (size_t)t * ROWS * D_DIM;
    const _Float16* WT = w16t + (size_t)t * WELEM;
    const float* bias = (t == 0) ? bq : ((t == 1) ? bk : bv);

    int wid = threadIdx.x >> 5;
    int m0 = (blockIdx.x * 8 + wid) * 16;
    int n0 = blockIdx.y * 16;

    v8f c = {};
    for (int k0 = 0; k0 < D_DIM; k0 += 32) {
        __builtin_prefetch(A + (size_t)m0 * D_DIM + k0 + 64, 0, 0);
        v16h a = load_a_f16(A + (size_t)m0 * D_DIM + k0, D_DIM);
        v16h b = load_bt_f16(WT + (size_t)n0 * D_DIM + k0, D_DIM);
        c = wmma_f16(a, b, c);
    }

    int lane = threadIdx.x & 31;
    int n = lane & 15;
    int col = n0 + n;
    int h = col >> 6, d = col & 63;
    float bb = bias[col];
    if (t < 2) {
        _Float16* dst = (t == 0) ? hq : hk;
#pragma unroll
        for (int j = 0; j < 8; ++j) {
            int row = m0 + j + (lane >> 4) * 8;
            int bi = row >> 11, si = row & (S_LEN - 1);
            dst[(((size_t)bi * H_NUM + h) * S_LEN + si) * DK + d] =
                (_Float16)(c[j] + bb);
        }
    } else {
#pragma unroll
        for (int j = 0; j < 8; ++j) {
            int row = m0 + j + (lane >> 4) * 8;
            int bi = row >> 11, si = row & (S_LEN - 1);
            vT[(((size_t)bi * H_NUM + h) * DK + d) * S_LEN + si] =
                (_Float16)(c[j] + bb);
        }
    }
}

// ---------------- Stage 4: attention ----------------
// block = 256 (8 waves); grid (S/16, B*H); dynamic LDS ~139KB (320KB WGP LDS)
__global__ __launch_bounds__(256) void attn_kernel(
        const _Float16* __restrict__ hq, const _Float16* __restrict__ hk,
        const _Float16* __restrict__ vT,
        const int* __restrict__ mask, const float* __restrict__ pos,
        float* __restrict__ attn_out, _Float16* __restrict__ xbuf) {
    extern __shared__ char smem[];
    float* sc    = (float*)smem;                       // [16][2048]
    float* part  = (float*)(smem + 16 * S_LEN * 4);    // [8][16][16]
    float* red   = part + 8 * 256;                     // [16][16]
    float* rowm  = red + 256;                          // [16]
    float* rsum  = rowm + 16;                          // [16]
    _Float16* qtile = (_Float16*)(rsum + 16);          // [16][64] = 2KB

    int bh = blockIdx.y;
    int b  = bh >> 3;
    int h  = bh & 7;
    int q0 = blockIdx.x * 16;
    const _Float16* Q  = hq + ((size_t)bh * S_LEN + q0) * DK;
    const _Float16* K  = hk + (size_t)bh * S_LEN * DK;
    const _Float16* VT = vT + (size_t)bh * DK * S_LEN;
    int wid  = threadIdx.x >> 5;
    int lane = threadIdx.x & 31;

    // --- async-copy the Q tile (16x64 f16) into LDS: 128 lanes x 16B ---
    if (threadIdx.x < 128) {
        unsigned ldsoff = (unsigned)(size_t)qtile + threadIdx.x * 16u;
        const _Float16* gp = Q + (size_t)(threadIdx.x >> 3) * DK +
                             (threadIdx.x & 7) * 8;
        unsigned long long ga = (unsigned long long)(size_t)gp;
        asm volatile("global_load_async_to_lds_b128 %0, %1, off"
                     :: "v"(ldsoff), "v"(ga) : "memory");
    }
    asm volatile("s_wait_asynccnt 0x0" ::: "memory");
    __syncthreads();

    v16h a0 = load_a_f16(qtile, DK);        // K-dim 0..31
    v16h a1 = load_a_f16(qtile + 32, DK);   // K-dim 32..63
    const float scale = 0.125f;             // 1/sqrt(64)

    // --- scores: each wave covers 256 key columns (16 tiles) ---
    for (int kt = 0; kt < 16; ++kt) {
        int kc0 = wid * 256 + kt * 16;
        v16h b0 = load_bt_f16(K + (size_t)kc0 * DK, DK);
        v16h b1 = load_bt_f16(K + (size_t)kc0 * DK + 32, DK);
        v8f c = {};
        c = wmma_f16(a0, b0, c);
        c = wmma_f16(a1, b1, c);
        int n = lane & 15;
        int col = kc0 + n;
#pragma unroll
        for (int j = 0; j < 8; ++j) {
            int m = j + (lane >> 4) * 8;
            int row = q0 + m;
            float s = c[j] * scale + pos[(size_t)row * S_LEN + col];
            if (mask[(size_t)row * S_LEN + col] == 0) s = -3.0e38f;
            sc[m * S_LEN + col] = s;
        }
    }
    __syncthreads();

    // --- row max (16 threads per row) ---
    int r = threadIdx.x >> 4;
    int cst = threadIdx.x & 15;
    float mx = -3.4e38f;
    for (int col = cst; col < S_LEN; col += 16) mx = fmaxf(mx, sc[r * S_LEN + col]);
    red[r * 16 + cst] = mx;
    __syncthreads();
    if (cst == 0) {
        float m2 = -3.4e38f;
        for (int i = 0; i < 16; ++i) m2 = fmaxf(m2, red[r * 16 + i]);
        rowm[r] = m2;
    }
    __syncthreads();
    // --- exp + row sum ---
    float rm = rowm[r];
    float sum = 0.f;
    for (int col = cst; col < S_LEN; col += 16) {
        float p = __expf(sc[r * S_LEN + col] - rm);
        sc[r * S_LEN + col] = p;
        sum += p;
    }
    red[r * 16 + cst] = sum;
    __syncthreads();
    if (cst == 0) {
        float s2 = 0.f;
        for (int i = 0; i < 16; ++i) s2 += red[r * 16 + i];
        rsum[r] = s2;
    }
    __syncthreads();

    // --- normalize; stream attn to global with NT float4 stores ---
    float* ao = attn_out + ((size_t)bh * S_LEN + q0) * S_LEN;
    for (int base = threadIdx.x * 4; base < 16 * S_LEN; base += 256 * 4) {
        int row = base >> 11;
        int col = base & (S_LEN - 1);
        float inv = 1.0f / rsum[row];
        v4f pv = *(v4f*)(sc + row * S_LEN + col);
        pv *= inv;
        *(v4f*)(sc + row * S_LEN + col) = pv;
        __builtin_nontemporal_store(pv, (v4f*)(ao + (size_t)row * S_LEN + col));
    }
    __syncthreads();

    // --- P @ V : wave = (khalf, ntile); 32 WMMAs over its K half ---
    int nt  = wid & 3;
    int khf = wid >> 2;
    v8f c = {};
    for (int k0 = khf * 1024; k0 < khf * 1024 + 1024; k0 += 32) {
        v16h a = load_a_f32(sc + k0, S_LEN);
        v16h bfrag = load_bt_f16(VT + (size_t)(nt * 16) * S_LEN + k0, S_LEN);
        c = wmma_f16(a, bfrag, c);
    }
    {
        int n = lane & 15;
#pragma unroll
        for (int j = 0; j < 8; ++j) {
            int m = j + (lane >> 4) * 8;
            part[(wid * 16 + m) * 16 + n] = c[j];
        }
    }
    __syncthreads();
    // reduce two K-halves, store x f16 [B,S,D]
    for (int idx = threadIdx.x; idx < 1024; idx += 256) {
        int nt2 = idx >> 8;
        int m   = (idx >> 4) & 15;
        int n   = idx & 15;
        float x = part[(nt2 * 16 + m) * 16 + n] + part[((nt2 + 4) * 16 + m) * 16 + n];
        xbuf[((size_t)b * S_LEN + q0 + m) * D_DIM + h * DK + nt2 * 16 + n] =
            (_Float16)x;
    }
}

// ---------------- Stage 5: output projection (WMMA) ----------------
__global__ __launch_bounds__(256) void gemm_out_kernel(
        const _Float16* __restrict__ xbuf, const _Float16* __restrict__ woT,
        const float* __restrict__ bo, const float* __restrict__ ls,
        float* __restrict__ out) {
    int wid = threadIdx.x >> 5;
    int m0 = (blockIdx.x * 8 + wid) * 16;
    int n0 = blockIdx.y * 16;
    v8f c = {};
    for (int k0 = 0; k0 < D_DIM; k0 += 32) {
        v16h a = load_a_f16(xbuf + (size_t)m0 * D_DIM + k0, D_DIM);
        v16h b = load_bt_f16(woT + (size_t)n0 * D_DIM + k0, D_DIM);
        c = wmma_f16(a, b, c);
    }
    int lane = threadIdx.x & 31;
    int n = lane & 15;
    int col = n0 + n;
    float bb = bo[col];
    float scl = ls[col];
#pragma unroll
    for (int j = 0; j < 8; ++j) {
        int row = m0 + j + (lane >> 4) * 8;
        __builtin_nontemporal_store((c[j] + bb) * scl,
                                    out + (size_t)row * D_DIM + col);
    }
}

// ---------------- launch ----------------
extern "C" void kernel_launch(void* const* d_in, const int* in_sizes, int n_in,
                              void* d_out, int out_size, void* d_ws, size_t ws_size,
                              hipStream_t stream) {
    const float* q     = (const float*)d_in[0];
    const float* k     = (const float*)d_in[1];
    const float* v     = (const float*)d_in[2];
    const int*   mask  = (const int*)d_in[3];
    const float* pos   = (const float*)d_in[4];
    const float* ln_g  = (const float*)d_in[5];
    const float* ln_b  = (const float*)d_in[6];
    const float* wq    = (const float*)d_in[7];
    const float* bq    = (const float*)d_in[8];
    const float* wk    = (const float*)d_in[9];
    const float* bk    = (const float*)d_in[10];
    const float* wv_w  = (const float*)d_in[11];
    const float* bv    = (const float*)d_in[12];
    const float* wo    = (const float*)d_in[13];
    const float* bo    = (const float*)d_in[14];
    const float* ls    = (const float*)d_in[15];

    char* ws = (char*)d_ws;
    size_t off = 0;
    _Float16* xn   = (_Float16*)(ws + off); off += (size_t)3 * ROWS * D_DIM * 2;
    _Float16* w16t = (_Float16*)(ws + off); off += (size_t)4 * WELEM * 2;
    _Float16* hq   = (_Float16*)(ws + off); off += (size_t)BATCH * H_NUM * S_LEN * DK * 2;
    _Float16* hk   = (_Float16*)(ws + off); off += (size_t)BATCH * H_NUM * S_LEN * DK * 2;
    _Float16* vT   = (_Float16*)(ws + off); off += (size_t)BATCH * H_NUM * S_LEN * DK * 2;
    _Float16* xbuf = (_Float16*)(ws + off); off += (size_t)ROWS * D_DIM * 2;

    float* out  = (float*)d_out;
    float* attn = out + (size_t)ROWS * D_DIM;

    const int SMEM = 16 * S_LEN * 4 + 8 * 256 * 4 + 256 * 4 + 16 * 4 + 16 * 4
                   + 16 * DK * 2;
    (void)hipFuncSetAttribute((const void*)attn_kernel,
                              hipFuncAttributeMaxDynamicSharedMemorySize, SMEM);

    ln_f16_kernel<<<dim3(ROWS / 4, 3), 128, 0, stream>>>(q, k, v, ln_g, ln_b, xn);
    wcvt_kernel<<<4096, 256, 0, stream>>>(wq, wk, wv_w, wo, w16t);
    gemm_qkv_kernel<<<dim3(64, 32, 3), 256, 0, stream>>>(xn, w16t, bq, bk, bv,
                                                         hq, hk, vT);
    attn_kernel<<<dim3(S_LEN / 16, BATCH * H_NUM), 256, SMEM, stream>>>(
        hq, hk, vT, mask, pos, attn, xbuf);
    gemm_out_kernel<<<dim3(64, 32), 256, 0, stream>>>(xbuf, w16t + 3 * WELEM,
                                                      bo, ls, out);
}
